// SegmentTransformer_34935263985878
// MI455X (gfx1250) — matrix operand order repaired and checked by reading, not compiled, not run on hardware
//
#include <hip/hip_runtime.h>
#include <hip/hip_bf16.h>

// ---------------- model dims ----------------
#define E_  512
#define H_  8
#define DH_ 64
#define NL_ 6
#define F_  1024
#define S_  512
#define L_  513
#define B_  32
#define LP_ 544            // L padded to multiple of 32 (and 16)
#define EPS_ 1e-5f
#define SCALE_ 0.125f      // 1/sqrt(64)

typedef __attribute__((ext_vector_type(16))) __bf16       v16bf;
typedef __attribute__((ext_vector_type(8)))  float        v8f;
typedef __attribute__((ext_vector_type(4)))  unsigned int u32x4;

union BfVec { v16bf v; u32x4 q[2]; };

// round-to-nearest-even f32 -> bf16 (stored as ushort)
__device__ __forceinline__ unsigned short f2bf(float f) {
    unsigned int u = __float_as_uint(f);
    if ((u & 0x7fffffffu) > 0x7f800000u) return (unsigned short)0x7fc0; // NaN
    u += 0x7fffu + ((u >> 16) & 1u);
    return (unsigned short)(u >> 16);
}

// A operand (16xK tile, row-major bf16, K-step of 32). base -> A[row0][kbase], ld in elems.
// ISA 7.12.2: lane<16 holds rows, elems 0..7 = K[hi*8..], elems 8..15 = K[hi*8+16..]
__device__ __forceinline__ v16bf load_a(const unsigned short* base, int ld, int lane) {
    int row = lane & 15, hi = lane >> 4;
    const unsigned short* p = base + row * ld + hi * 8;
    BfVec u;
    u.q[0] = *(const u32x4*)(p);
    u.q[1] = *(const u32x4*)(p + 16);
    return u.v;
}

// B operand (Kx16 tile) sourced from an N-major matrix (row n holds K contiguous).
// base -> Bn[n0][kbase]. col = lane&15: elems 0..7 = K[hi*8..], elems 8..15 = K[hi*8+16..]
__device__ __forceinline__ v16bf load_b(const unsigned short* base, int ld, int lane) {
    int col = lane & 15, hi = lane >> 4;
    const unsigned short* p = base + col * ld + hi * 8;
    BfVec u;
    u.q[0] = *(const u32x4*)(p);
    u.q[1] = *(const u32x4*)(p + 16);
    return u.v;
}

#define WMMA_BF16(a, b, c) \
    __builtin_amdgcn_wmma_f32_16x16x32_bf16(false, (a), false, (b), (short)0, (c), false, false)

__device__ __forceinline__ void mma8(v8f acc[2][4], v16bf a0, v16bf a1,
                                     v16bf b0, v16bf b1, v16bf b2, v16bf b3) {
    acc[0][0] = WMMA_BF16(a0, b0, acc[0][0]);
    acc[1][0] = WMMA_BF16(a1, b0, acc[1][0]);
    acc[0][1] = WMMA_BF16(a0, b1, acc[0][1]);
    acc[1][1] = WMMA_BF16(a1, b1, acc[1][1]);
    acc[0][2] = WMMA_BF16(a0, b2, acc[0][2]);
    acc[1][2] = WMMA_BF16(a1, b2, acc[1][2]);
    acc[0][3] = WMMA_BF16(a0, b3, acc[0][3]);
    acc[1][3] = WMMA_BF16(a1, b3, acc[1][3]);
}

__device__ __forceinline__ void mma4(v8f acc[4], v16bf a,
                                     v16bf b0, v16bf b1, v16bf b2, v16bf b3) {
    acc[0] = WMMA_BF16(a, b0, acc[0]);
    acc[1] = WMMA_BF16(a, b1, acc[1]);
    acc[2] = WMMA_BF16(a, b2, acc[2]);
    acc[3] = WMMA_BF16(a, b3, acc[3]);
}

__device__ __forceinline__ float wave_max(float v) {
    for (int o = 16; o > 0; o >>= 1) v = fmaxf(v, __shfl_xor(v, o, 32));
    return v;
}
__device__ __forceinline__ float wave_sum(float v) {
    for (int o = 16; o > 0; o >>= 1) v += __shfl_xor(v, o, 32);
    return v;
}

// ---------------- embed: src = tok_w * x_num + tok_b (+pad), plus bf16 and transposed-bf16 copies
__global__ void embed_kernel(const float* __restrict__ x,
                             const float* __restrict__ tok_w,
                             const float* __restrict__ tok_b,
                             float* __restrict__ src,
                             unsigned short* __restrict__ src_bf,
                             unsigned short* __restrict__ srcT) {
    int idx = blockIdx.x * blockDim.x + threadIdx.x;
    if (idx >= B_ * LP_ * E_) return;
    int e = idx % E_;
    int l = (idx / E_) % LP_;
    int b = idx / (E_ * LP_);
    float v = 0.f;
    if (l < L_) {
        float xn = (l == 0) ? 1.f : x[b * S_ + (l - 1)];
        v = tok_w[l * E_ + e] * xn + tok_b[l * E_ + e];
    }
    src[idx] = v;
    unsigned short bv = f2bf(v);
    src_bf[idx] = bv;
    srcT[((size_t)b * E_ + e) * LP_ + l] = bv;
}

// ---------------- fp32 -> bf16 weight conversion (grid-stride)
__global__ void cvt_kernel(const float* __restrict__ in, unsigned short* __restrict__ out, int n) {
    for (int i = blockIdx.x * blockDim.x + threadIdx.x; i < n; i += gridDim.x * blockDim.x)
        out[i] = f2bf(in[i]);
}

// ---------------- Q/K projection: out[m,n] = sum_k src_bf[m,k]*W[n,k] + bias[n] (bf16 out)
// 32x64 output tile per wave; explicit double-buffered operands so WMMAs never
// wait on the loads of their own k-step.
// grid = (E/64, B*LP/32), block = 32
__global__ __launch_bounds__(32)
void gemm_qk_kernel(const unsigned short* __restrict__ A,
                    const unsigned short* __restrict__ Bn,   // N-major: (E, E)
                    const float* __restrict__ bias,
                    unsigned short* __restrict__ out) {
    int lane = threadIdx.x;
    int n0 = blockIdx.x * 64;
    int m0 = blockIdx.y * 32;
    v8f z = {};
    v8f acc[2][4];
    for (int i = 0; i < 2; ++i)
        for (int j = 0; j < 4; ++j) acc[i][j] = z;
    const unsigned short* ap0 = A + (size_t)m0 * E_;
    const unsigned short* ap1 = ap0 + (size_t)16 * E_;
    const unsigned short* bp  = Bn + (size_t)n0 * E_;

    v16bf a0 = load_a(ap0, E_, lane);
    v16bf a1 = load_a(ap1, E_, lane);
    v16bf b0 = load_b(bp,                    E_, lane);
    v16bf b1 = load_b(bp + (size_t)16 * E_,  E_, lane);
    v16bf b2 = load_b(bp + (size_t)32 * E_,  E_, lane);
    v16bf b3 = load_b(bp + (size_t)48 * E_,  E_, lane);

    for (int k = 0; k < E_; k += 32) {
        v16bf ca0 = a0, ca1 = a1, cb0 = b0, cb1 = b1, cb2 = b2, cb3 = b3;
        if (k + 32 < E_) {
            int kn = k + 32;
            a0 = load_a(ap0 + kn, E_, lane);
            a1 = load_a(ap1 + kn, E_, lane);
            b0 = load_b(bp + kn,                   E_, lane);
            b1 = load_b(bp + (size_t)16 * E_ + kn, E_, lane);
            b2 = load_b(bp + (size_t)32 * E_ + kn, E_, lane);
            b3 = load_b(bp + (size_t)48 * E_ + kn, E_, lane);
        }
        mma8(acc, ca0, ca1, cb0, cb1, cb2, cb3);
    }

    int col = lane & 15, hi = lane >> 4;
    for (int j = 0; j < 4; ++j) {
        float bn = bias[n0 + j * 16 + col];
        for (int i = 0; i < 2; ++i)
            for (int r = 0; r < 8; ++r)
                out[(size_t)(m0 + i * 16 + r + hi * 8) * E_ + n0 + j * 16 + col] =
                    f2bf(acc[i][j][r] + bn);
    }
}

// ---------------- fused attention for a 16-row tile of one batch
// grid = B * (LP/16), block = 32, dyn smem = 2*16*LP*4 + 16*LP*2
__global__ __launch_bounds__(32)
void attn_kernel(const unsigned short* __restrict__ q,
                 const unsigned short* __restrict__ k,
                 const unsigned short* __restrict__ srcT,
                 const float* __restrict__ bias,
                 const float* __restrict__ lnw,
                 const float* __restrict__ lnb,
                 float* __restrict__ src,
                 unsigned short* __restrict__ src_bf) {
    extern __shared__ char smem[];
    float* s_sc = (float*)smem;                              // 16 x LP
    float* s_avg = s_sc + 16 * LP_;                          // 16 x LP
    unsigned short* s_w2 = (unsigned short*)(s_avg + 16 * LP_); // 16 x LP bf16

    int lane = threadIdx.x;
    int b = blockIdx.x / (LP_ / 16);
    int t = blockIdx.x % (LP_ / 16);
    int row0 = t * 16;
    int col = lane & 15, hi = lane >> 4;

    for (int i = lane; i < 16 * LP_; i += 32) s_avg[i] = 0.f;
    __syncthreads();

    const unsigned short* qb = q + (size_t)b * LP_ * E_;
    const unsigned short* kb = k + (size_t)b * LP_ * E_;

    for (int h = 0; h < H_; ++h) {
        int hoff = h * DH_;
        v16bf a0 = load_a(qb + (size_t)row0 * E_ + hoff,      E_, lane);
        v16bf a1 = load_a(qb + (size_t)row0 * E_ + hoff + 32, E_, lane);
        // double-buffered B across the 34 n-tiles
        v16bf b0 = load_b(kb + hoff,      E_, lane);
        v16bf b1 = load_b(kb + hoff + 32, E_, lane);
        for (int nt = 0; nt < LP_ / 16; ++nt) {
            v16bf cb0 = b0, cb1 = b1;
            if (nt + 1 < LP_ / 16) {
                const unsigned short* nb = kb + (size_t)(nt + 1) * 16 * E_ + hoff;
                b0 = load_b(nb,      E_, lane);
                b1 = load_b(nb + 32, E_, lane);
            }
            v8f acc = {};
            acc = WMMA_BF16(a0, cb0, acc);
            acc = WMMA_BF16(a1, cb1, acc);
            int n0 = nt * 16;
            for (int r = 0; r < 8; ++r)
                s_sc[(r + hi * 8) * LP_ + n0 + col] = acc[r] * SCALE_;
        }
        __syncthreads();
        // per-row softmax over valid m<L, accumulate mean over heads
        for (int r = 0; r < 16; ++r) {
            float mx = -1e30f;
            for (int m = lane; m < L_; m += 32) mx = fmaxf(mx, s_sc[r * LP_ + m]);
            mx = wave_max(mx);
            float sum = 0.f;
            for (int m = lane; m < L_; m += 32) {
                float e = __expf(s_sc[r * LP_ + m] - mx);
                s_sc[r * LP_ + m] = e;
                sum += e;
            }
            sum = wave_sum(sum);
            float inv = (1.0f / H_) / sum;
            for (int m = lane; m < L_; m += 32) s_avg[r * LP_ + m] += s_sc[r * LP_ + m] * inv;
        }
        __syncthreads();
    }

    // w2 = softmax(avg + bias) -> bf16 in LDS (zero pad m >= L)
    for (int r = 0; r < 16; ++r) {
        int lg = row0 + r;
        const float* brow = bias + (size_t)((lg < L_) ? lg : (L_ - 1)) * L_;
        float mx = -1e30f;
        for (int m = lane; m < L_; m += 32) {
            float v = s_avg[r * LP_ + m] + brow[m];
            s_avg[r * LP_ + m] = v;
            mx = fmaxf(mx, v);
        }
        mx = wave_max(mx);
        float sum = 0.f;
        for (int m = lane; m < L_; m += 32) {
            float e = __expf(s_avg[r * LP_ + m] - mx);
            s_avg[r * LP_ + m] = e;
            sum += e;
        }
        sum = wave_sum(sum);
        float inv = 1.f / sum;
        for (int m = lane; m < LP_; m += 32)
            s_w2[r * LP_ + m] = (m < L_) ? f2bf(s_avg[r * LP_ + m] * inv) : (unsigned short)0;
    }
    __syncthreads();

    // attn_out = w2 @ src  (B from transposed src; 16x64 tile, double-buffered operands)
    const unsigned short* sT = srcT + (size_t)b * E_ * LP_;
    for (int nt = 0; nt < E_ / 64; ++nt) {
        int n0 = nt * 64;
        v8f z = {};
        v8f acc[4];
        for (int j = 0; j < 4; ++j) acc[j] = z;
        const unsigned short* bp = sT + (size_t)n0 * LP_;
        v16bf a  = load_a(s_w2, LP_, lane);
        v16bf b0 = load_b(bp,                     LP_, lane);
        v16bf b1 = load_b(bp + (size_t)16 * LP_,  LP_, lane);
        v16bf b2 = load_b(bp + (size_t)32 * LP_,  LP_, lane);
        v16bf b3 = load_b(bp + (size_t)48 * LP_,  LP_, lane);
        for (int kk = 0; kk < LP_; kk += 32) {
            v16bf ca = a, cb0 = b0, cb1 = b1, cb2 = b2, cb3 = b3;
            if (kk + 32 < LP_) {
                int kn = kk + 32;
                a  = load_a(s_w2 + kn, LP_, lane);
                b0 = load_b(bp + kn,                    LP_, lane);
                b1 = load_b(bp + (size_t)16 * LP_ + kn, LP_, lane);
                b2 = load_b(bp + (size_t)32 * LP_ + kn, LP_, lane);
                b3 = load_b(bp + (size_t)48 * LP_ + kn, LP_, lane);
            }
            mma4(acc, ca, cb0, cb1, cb2, cb3);
        }
        for (int j = 0; j < 4; ++j)
            for (int r = 0; r < 8; ++r)
                s_sc[(r + hi * 8) * E_ + n0 + j * 16 + col] = acc[j][r]; // reuse s_sc, stride E
    }
    __syncthreads();

    // residual + LayerNorm, write src (f32) and src_bf
    for (int r = 0; r < 16; ++r) {
        int lg = row0 + r;
        size_t rowoff = (size_t)(b * LP_ + lg) * E_;
        if (lg >= L_) {
            for (int e = lane; e < E_; e += 32) { src[rowoff + e] = 0.f; src_bf[rowoff + e] = 0; }
            continue;
        }
        float vv[16];
        float s = 0.f; int j = 0;
        for (int e = lane; e < E_; e += 32) {
            float v = src[rowoff + e] + s_sc[r * E_ + e];
            vv[j++] = v; s += v;
        }
        float mean = wave_sum(s) * (1.f / E_);
        float vs = 0.f;
        for (int i = 0; i < 16; ++i) { float d = vv[i] - mean; vs += d * d; }
        float inv = rsqrtf(wave_sum(vs) * (1.f / E_) + EPS_);
        j = 0;
        for (int e = lane; e < E_; e += 32) {
            float y = (vv[j++] - mean) * inv * lnw[e] + lnb[e];
            src[rowoff + e] = y;
            src_bf[rowoff + e] = f2bf(y);
        }
    }
}

// ---------------- fused FFN + LN for a 16-row tile
// grid = B * (LP/16), block = 32, dyn smem = 16*F*2 + 16*E*4
__global__ __launch_bounds__(32)
void ffn_kernel(const unsigned short* __restrict__ fc1,   // N-major (F, E) bf16
                const float* __restrict__ fc1_b,
                const unsigned short* __restrict__ fc2,   // N-major (E, F) bf16
                const float* __restrict__ fc2_b,
                const float* __restrict__ lnw,
                const float* __restrict__ lnb,
                float* __restrict__ src,
                unsigned short* __restrict__ src_bf,
                unsigned short* __restrict__ srcT) {
    extern __shared__ char smem[];
    unsigned short* s_hid = (unsigned short*)smem;     // 16 x F bf16
    float* s_out = (float*)(smem + 16 * F_ * 2);       // 16 x E f32

    int lane = threadIdx.x;
    int b = blockIdx.x / (LP_ / 16);
    int t = blockIdx.x % (LP_ / 16);
    int row0 = t * 16;
    int col = lane & 15, hi = lane >> 4;

    const unsigned short* Arow = src_bf + (size_t)(b * LP_ + row0) * E_;

    // GEMM1 + ReLU -> LDS hidden (bf16); 16x64 tiles, double-buffered operands
    for (int nt = 0; nt < F_ / 64; ++nt) {
        int n0 = nt * 64;
        v8f z = {};
        v8f acc[4];
        for (int j = 0; j < 4; ++j) acc[j] = z;
        const unsigned short* bp = fc1 + (size_t)n0 * E_;
        v16bf a  = load_a(Arow, E_, lane);
        v16bf b0 = load_b(bp,                    E_, lane);
        v16bf b1 = load_b(bp + (size_t)16 * E_,  E_, lane);
        v16bf b2 = load_b(bp + (size_t)32 * E_,  E_, lane);
        v16bf b3 = load_b(bp + (size_t)48 * E_,  E_, lane);
        for (int k = 0; k < E_; k += 32) {
            v16bf ca = a, cb0 = b0, cb1 = b1, cb2 = b2, cb3 = b3;
            if (k + 32 < E_) {
                int kn = k + 32;
                a  = load_a(Arow + kn, E_, lane);
                b0 = load_b(bp + kn,                   E_, lane);
                b1 = load_b(bp + (size_t)16 * E_ + kn, E_, lane);
                b2 = load_b(bp + (size_t)32 * E_ + kn, E_, lane);
                b3 = load_b(bp + (size_t)48 * E_ + kn, E_, lane);
            }
            mma4(acc, ca, cb0, cb1, cb2, cb3);
        }
        for (int j = 0; j < 4; ++j) {
            float bn = fc1_b[n0 + j * 16 + col];
            for (int r = 0; r < 8; ++r)
                s_hid[(r + hi * 8) * F_ + n0 + j * 16 + col] = f2bf(fmaxf(acc[j][r] + bn, 0.f));
        }
    }
    __syncthreads();

    // GEMM2 -> LDS out (f32); 16x64 tiles, double-buffered operands
    for (int nt = 0; nt < E_ / 64; ++nt) {
        int n0 = nt * 64;
        v8f z = {};
        v8f acc[4];
        for (int j = 0; j < 4; ++j) acc[j] = z;
        const unsigned short* bp = fc2 + (size_t)n0 * F_;
        v16bf a  = load_a(s_hid, F_, lane);
        v16bf b0 = load_b(bp,                    F_, lane);
        v16bf b1 = load_b(bp + (size_t)16 * F_,  F_, lane);
        v16bf b2 = load_b(bp + (size_t)32 * F_,  F_, lane);
        v16bf b3 = load_b(bp + (size_t)48 * F_,  F_, lane);
        for (int k = 0; k < F_; k += 32) {
            v16bf ca = a, cb0 = b0, cb1 = b1, cb2 = b2, cb3 = b3;
            if (k + 32 < F_) {
                int kn = k + 32;
                a  = load_a(s_hid + kn, F_, lane);
                b0 = load_b(bp + kn,                   F_, lane);
                b1 = load_b(bp + (size_t)16 * F_ + kn, F_, lane);
                b2 = load_b(bp + (size_t)32 * F_ + kn, F_, lane);
                b3 = load_b(bp + (size_t)48 * F_ + kn, F_, lane);
            }
            mma4(acc, ca, cb0, cb1, cb2, cb3);
        }
        for (int j = 0; j < 4; ++j) {
            float bn = fc2_b[n0 + j * 16 + col];
            for (int r = 0; r < 8; ++r)
                s_out[(r + hi * 8) * E_ + n0 + j * 16 + col] = acc[j][r] + bn;
        }
    }
    __syncthreads();

    // residual + LN, write src trio (incl. transposed bf16 for next layer's attention)
    for (int r = 0; r < 16; ++r) {
        int lg = row0 + r;
        size_t rowoff = (size_t)(b * LP_ + lg) * E_;
        if (lg >= L_) {
            for (int e = lane; e < E_; e += 32) {
                src[rowoff + e] = 0.f;
                src_bf[rowoff + e] = 0;
                srcT[((size_t)b * E_ + e) * LP_ + lg] = 0;
            }
            continue;
        }
        float vv[16];
        float s = 0.f; int j = 0;
        for (int e = lane; e < E_; e += 32) {
            float v = src[rowoff + e] + s_out[r * E_ + e];
            vv[j++] = v; s += v;
        }
        float mean = wave_sum(s) * (1.f / E_);
        float vs = 0.f;
        for (int i = 0; i < 16; ++i) { float d = vv[i] - mean; vs += d * d; }
        float inv = rsqrtf(wave_sum(vs) * (1.f / E_) + EPS_);
        j = 0;
        for (int e = lane; e < E_; e += 32) {
            float y = (vv[j++] - mean) * inv * lnw[e] + lnb[e];
            unsigned short yb = f2bf(y);
            src[rowoff + e] = y;
            src_bf[rowoff + e] = yb;
            srcT[((size_t)b * E_ + e) * LP_ + lg] = yb;
        }
    }
}

// ---------------- output: src[:, 0, :]
__global__ void out_kernel(const float* __restrict__ src, float* __restrict__ out) {
    int idx = blockIdx.x * blockDim.x + threadIdx.x;
    if (idx >= B_ * E_) return;
    int b = idx / E_, e = idx % E_;
    out[idx] = src[(size_t)(b * LP_) * E_ + e];
}

extern "C" void kernel_launch(void* const* d_in, const int* in_sizes, int n_in,
                              void* d_out, int out_size, void* d_ws, size_t ws_size,
                              hipStream_t stream) {
    const float* x         = (const float*)d_in[0];
    const float* tok_w     = (const float*)d_in[1];
    const float* tok_b     = (const float*)d_in[2];
    const float* bias      = (const float*)d_in[3];
    const float* in_proj_w = (const float*)d_in[4];
    const float* in_proj_b = (const float*)d_in[5];
    const float* fc1_w     = (const float*)d_in[6];
    const float* fc1_b     = (const float*)d_in[7];
    const float* fc2_w     = (const float*)d_in[8];
    const float* fc2_b     = (const float*)d_in[9];
    const float* ln1_w     = (const float*)d_in[10];
    const float* ln1_b     = (const float*)d_in[11];
    const float* ln2_w     = (const float*)d_in[12];
    const float* ln2_b     = (const float*)d_in[13];

    // workspace partition
    char* p = (char*)d_ws;
    const size_t nAct = (size_t)B_ * LP_ * E_;
    float*          src    = (float*)p;          p += nAct * sizeof(float);
    unsigned short* src_bf = (unsigned short*)p; p += nAct * sizeof(unsigned short);
    unsigned short* srcT   = (unsigned short*)p; p += nAct * sizeof(unsigned short);
    unsigned short* q_bf   = (unsigned short*)p; p += nAct * sizeof(unsigned short);
    unsigned short* k_bf   = (unsigned short*)p; p += nAct * sizeof(unsigned short);
    unsigned short* wqk_bf = (unsigned short*)p; p += (size_t)NL_ * 2 * E_ * E_ * 2;
    unsigned short* fc1_bf = (unsigned short*)p; p += (size_t)NL_ * F_ * E_ * 2;
    unsigned short* fc2_bf = (unsigned short*)p; p += (size_t)NL_ * E_ * F_ * 2;
    (void)ws_size; (void)in_sizes; (void)n_in; (void)out_size;

    // embed + padding
    {
        int n = B_ * LP_ * E_;
        embed_kernel<<<(n + 255) / 256, 256, 0, stream>>>(x, tok_w, tok_b, src, src_bf, srcT);
    }
    // weight conversion to bf16 (q/k rows of in_proj, fc1, fc2)
    for (int i = 0; i < NL_; ++i) {
        int n = 2 * E_ * E_;
        cvt_kernel<<<1024, 256, 0, stream>>>(in_proj_w + (size_t)i * 3 * E_ * E_,
                                             wqk_bf + (size_t)i * 2 * E_ * E_, n);
    }
    cvt_kernel<<<4096, 256, 0, stream>>>(fc1_w, fc1_bf, NL_ * F_ * E_);
    cvt_kernel<<<4096, 256, 0, stream>>>(fc2_w, fc2_bf, NL_ * E_ * F_);

    const int mtiles16 = (B_ * LP_) / 16;               // 1088
    const dim3 gGemm(E_ / 64, (B_ * LP_) / 32);         // (8, 544)
    const int attn_smem = 2 * 16 * LP_ * 4 + 16 * LP_ * 2;  // 87,040 B
    const int ffn_smem  = 16 * F_ * 2 + 16 * E_ * 4;        // 65,536 B

    for (int i = 0; i < NL_; ++i) {
        const unsigned short* wq = wqk_bf + (size_t)i * 2 * E_ * E_;
        const unsigned short* wk = wq + (size_t)E_ * E_;
        const float* bq = in_proj_b + (size_t)i * 3 * E_;
        const float* bk = bq + E_;

        gemm_qk_kernel<<<gGemm, 32, 0, stream>>>(src_bf, wq, bq, q_bf);
        gemm_qk_kernel<<<gGemm, 32, 0, stream>>>(src_bf, wk, bk, k_bf);

        attn_kernel<<<mtiles16, 32, attn_smem, stream>>>(
            q_bf, k_bf, srcT, bias, ln1_w + (size_t)i * E_, ln1_b + (size_t)i * E_,
            src, src_bf);

        ffn_kernel<<<mtiles16, 32, ffn_smem, stream>>>(
            fc1_bf + (size_t)i * F_ * E_, fc1_b + (size_t)i * F_,
            fc2_bf + (size_t)i * E_ * F_, fc2_b + (size_t)i * E_,
            ln2_w + (size_t)i * E_, ln2_b + (size_t)i * E_,
            src, src_bf, srcT);
    }

    out_kernel<<<(B_ * E_ + 255) / 256, 256, 0, stream>>>(src, (float*)d_out);
}
